// EdgeConv_17145509446484
// MI455X (gfx1250) — matrix operand order repaired and checked by reading, not compiled, not run on hardware
//
#include <hip/hip_runtime.h>

#define KNN   32
#define NPTS  4096
#define NB    8
#define OC    64
#define EPSF  1e-5f
#define SLOPE 0.2f
#define APAD  72          // row pitch in f16 for LDS tiles (36 dwords -> conflict-free)
#define PPB   8           // points per block (one wave each)

typedef __attribute__((ext_vector_type(8)))  _Float16 v8h;
typedef __attribute__((ext_vector_type(16))) _Float16 v16h;
typedef __attribute__((ext_vector_type(8)))  float    v8f;

__device__ __forceinline__ v16h cat8(v8h a, v8h b) {
    return __builtin_shufflevector(a, b, 0,1,2,3,4,5,6,7,8,9,10,11,12,13,14,15);
}

// ---------------------------------------------------------------------------
// Kernel 1: kNN (feature-space, neg squared distance, top-32 incl. self)
// One block per (b,n) row. dist cached in LDS, 32 iterative arg-max passes.
// ---------------------------------------------------------------------------
__global__ void __launch_bounds__(256)
knn_kernel(const float* __restrict__ x, int* __restrict__ idxout) {
    __shared__ float dist[NPTS];
    __shared__ float rv[8];
    __shared__ int   ri[8];

    const int row = blockIdx.x;             // 0..NB*NPTS-1
    const int b   = row >> 12;
    const int n   = row & (NPTS - 1);
    const float* xb = x + (size_t)b * 3 * NPTS;

    const float c0 = xb[n], c1 = xb[NPTS + n], c2 = xb[2 * NPTS + n];
    const float x2n = c0 * c0 + c1 * c1 + c2 * c2;

    const int tid = threadIdx.x;
    for (int m = tid; m < NPTS; m += 256) {
        float a0 = xb[m], a1 = xb[NPTS + m], a2 = xb[2 * NPTS + m];
        float inner = c0 * a0 + c1 * a1 + c2 * a2;
        float x2m = a0 * a0 + a1 * a1 + a2 * a2;
        dist[m] = 2.0f * inner - x2n - x2m;
    }
    __syncthreads();

    const int lane = tid & 31, wv = tid >> 5;
    for (int j = 0; j < KNN; ++j) {
        float bv = -3.4e38f; int bi = 0x7fffffff;
        for (int m = tid; m < NPTS; m += 256) {
            float v = dist[m];
            if (v > bv) { bv = v; bi = m; }          // ascending scan -> lowest idx on tie
        }
        #pragma unroll
        for (int off = 16; off; off >>= 1) {
            float ov = __shfl_xor(bv, off, 32);
            int   oi = __shfl_xor(bi, off, 32);
            if (ov > bv || (ov == bv && oi < bi)) { bv = ov; bi = oi; }
        }
        if (lane == 0) { rv[wv] = bv; ri[wv] = bi; }
        __syncthreads();
        if (tid == 0) {
            float Bv = rv[0]; int Bi = ri[0];
            #pragma unroll
            for (int t = 1; t < 8; ++t)
                if (rv[t] > Bv || (rv[t] == Bv && ri[t] < Bi)) { Bv = rv[t]; Bi = ri[t]; }
            idxout[(size_t)row * KNN + j] = Bi;
            dist[Bi] = -3.4e38f;
        }
        __syncthreads();
    }
}

// ---------------------------------------------------------------------------
// Kernel 2: first & second moments of the 6-dim edge vectors (BN1 is linear
// in these: mean(h1)=W1*mean(e), E[h1^2]=W1^T E[ee^T] W1).
// ---------------------------------------------------------------------------
__global__ void __launch_bounds__(256)
moments_kernel(const float* __restrict__ x, const int* __restrict__ idx,
               float* __restrict__ mom) {
    float acc[42];
    #pragma unroll
    for (int i = 0; i < 42; ++i) acc[i] = 0.f;

    const int total = NB * NPTS * KNN;
    const int stride = gridDim.x * blockDim.x;
    for (int e = blockIdx.x * blockDim.x + threadIdx.x; e < total; e += stride) {
        int b = e >> 17;                        // NPTS*KNN = 2^17
        int rem = e & (NPTS * KNN - 1);
        int n = rem >> 5;
        int j = idx[e];
        const float* xb = x + (size_t)b * 3 * NPTS;
        float ee[6];
        float c0 = xb[n], c1 = xb[NPTS + n], c2 = xb[2 * NPTS + n];
        ee[0] = c0; ee[1] = c1; ee[2] = c2;
        ee[3] = xb[j] - c0; ee[4] = xb[NPTS + j] - c1; ee[5] = xb[2 * NPTS + j] - c2;
        #pragma unroll
        for (int c = 0; c < 6; ++c) {
            acc[c] += ee[c];
            #pragma unroll
            for (int d = 0; d < 6; ++d) acc[6 + c * 6 + d] += ee[c] * ee[d];
        }
    }
    __shared__ float sacc[42];
    if (threadIdx.x < 42) sacc[threadIdx.x] = 0.f;
    __syncthreads();
    #pragma unroll
    for (int i = 0; i < 42; ++i) atomicAdd(&sacc[i], acc[i]);
    __syncthreads();
    if (threadIdx.x < 42) atomicAdd(&mom[threadIdx.x], sacc[threadIdx.x]);
}

// ---------------------------------------------------------------------------
// Kernel 3: BN1 -> fused scale/bias per output channel
// ---------------------------------------------------------------------------
__global__ void finalize1_kernel(const float* __restrict__ W1,
                                 const float* __restrict__ gamma1,
                                 const float* __restrict__ beta1,
                                 const float* __restrict__ mom,
                                 float* __restrict__ scale1, float* __restrict__ bias1) {
    int o = threadIdx.x;
    if (o >= OC) return;
    const float inv = 1.0f / (float)(NB * NPTS * KNN);
    float w[6];
    #pragma unroll
    for (int c = 0; c < 6; ++c) w[c] = W1[o * 6 + c];
    float mean = 0.f;
    #pragma unroll
    for (int c = 0; c < 6; ++c) mean += w[c] * mom[c] * inv;
    float ex2 = 0.f;
    #pragma unroll
    for (int c = 0; c < 6; ++c)
        #pragma unroll
        for (int d = 0; d < 6; ++d) ex2 += w[c] * w[d] * mom[6 + c * 6 + d] * inv;
    float var = ex2 - mean * mean;
    float sc = gamma1[o] * rsqrtf(var + EPSF);
    scale1[o] = sc;
    bias1[o] = beta1[o] - mean * sc;
}

// ---------------------------------------------------------------------------
// Shared per-wave stage: build H1n (32 x 64, f16, post BN1+LReLU) in LDS.
// Lane = neighbor k; 6->64 conv is 384 VALU MACs per lane.
// ---------------------------------------------------------------------------
__device__ __forceinline__ void compute_h1_tile(int p, int lane,
                                                const float* __restrict__ x,
                                                const int* __restrict__ idx,
                                                const float* sW1, const float* sS1,
                                                const float* sB1, _Float16* arow) {
    const int b = p >> 12;
    const int n = p & (NPTS - 1);
    const float* xb = x + (size_t)b * 3 * NPTS;
    const int j = idx[(size_t)p * KNN + lane];
    float c0 = xb[n], c1 = xb[NPTS + n], c2 = xb[2 * NPTS + n];
    float e3 = xb[j] - c0, e4 = xb[NPTS + j] - c1, e5 = xb[2 * NPTS + j] - c2;
    #pragma unroll 8
    for (int o = 0; o < OC; ++o) {
        const float* wr = &sW1[o * 6];
        float h = wr[0] * c0 + wr[1] * c1 + wr[2] * c2 + wr[3] * e3 + wr[4] * e4 + wr[5] * e5;
        h = h * sS1[o] + sB1[o];
        h = (h >= 0.f) ? h : SLOPE * h;
        arow[o] = (_Float16)h;
    }
}

// Per-wave WMMA: acc[mtile 0..1][ntile 0..3] += A(32x64) * W2^T(64x64)
#define WMMA_CORE(SA_WAVE, SW2, ACC, L16, HI)                                          \
    do {                                                                               \
        _Pragma("unroll")                                                              \
        for (int kc = 0; kc < 2; ++kc) {                                               \
            const int c0 = kc * 32;                                                    \
            v16h af[2];                                                                \
            _Pragma("unroll")                                                          \
            for (int t = 0; t < 2; ++t) {                                              \
                const _Float16* rp = &SA_WAVE[t * 16 + L16][0];                        \
                const int cb = c0 + (HI ? 8 : 0);                                      \
                af[t] = cat8(*(const v8h*)(rp + cb), *(const v8h*)(rp + cb + 16));     \
            }                                                                          \
            _Pragma("unroll")                                                          \
            for (int nt = 0; nt < 4; ++nt) {                                           \
                const _Float16* wp = &SW2[nt * 16 + L16][0];                           \
                const int cb = c0 + (HI ? 8 : 0);                                      \
                v16h bf = cat8(*(const v8h*)(wp + cb), *(const v8h*)(wp + cb + 16));   \
                ACC[0][nt] = __builtin_amdgcn_wmma_f32_16x16x32_f16(                   \
                    false, af[0], false, bf, (short)0, ACC[0][nt], false, false);      \
                ACC[1][nt] = __builtin_amdgcn_wmma_f32_16x16x32_f16(                   \
                    false, af[1], false, bf, (short)0, ACC[1][nt], false, false);      \
            }                                                                          \
        }                                                                              \
    } while (0)

// ---------------------------------------------------------------------------
// Kernel 4: pass over all edges computing h2 via WMMA; accumulate per-channel
// sum & sumsq for BN2 statistics.
// ---------------------------------------------------------------------------
__global__ void __launch_bounds__(256)
stats2_kernel(const float* __restrict__ x, const int* __restrict__ idx,
              const float* __restrict__ W1f, const float* __restrict__ W2f,
              const float* __restrict__ scale1, const float* __restrict__ bias1,
              float* __restrict__ gsum, float* __restrict__ gsq) {
    __shared__ _Float16 sW2[OC][APAD];
    __shared__ _Float16 sA[PPB][KNN][APAD];
    __shared__ float sW1[OC * 6];
    __shared__ float sS1[OC], sB1[OC];
    __shared__ float sSum[OC], sSq[OC];

    const int tid = threadIdx.x;
    for (int i = tid; i < OC * OC; i += 256) sW2[i >> 6][i & 63] = (_Float16)W2f[i];
    for (int i = tid; i < OC * 6; i += 256) sW1[i] = W1f[i];
    if (tid < OC) { sS1[tid] = scale1[tid]; sB1[tid] = bias1[tid]; sSum[tid] = 0.f; sSq[tid] = 0.f; }
    __syncthreads();

    const int w = tid >> 5, lane = tid & 31;
    const int p = blockIdx.x * PPB + w;
    compute_h1_tile(p, lane, x, idx, sW1, sS1, sB1, &sA[w][lane][0]);
    __syncthreads();

    const int l16 = lane & 15;
    const bool hi = lane >= 16;
    v8f acc[2][4] = {};
    WMMA_CORE(sA[w], sW2, acc, l16, hi);

    #pragma unroll
    for (int nt = 0; nt < 4; ++nt) {
        const int o = nt * 16 + l16;
        float s = 0.f, q = 0.f;
        #pragma unroll
        for (int i = 0; i < 8; ++i) {
            float v0 = acc[0][nt][i], v1 = acc[1][nt][i];
            s += v0 + v1;
            q += v0 * v0 + v1 * v1;
        }
        atomicAdd(&sSum[o], s);
        atomicAdd(&sSq[o], q);
    }
    __syncthreads();
    if (tid < OC) { atomicAdd(&gsum[tid], sSum[tid]); atomicAdd(&gsq[tid], sSq[tid]); }
}

// ---------------------------------------------------------------------------
// Kernel 5: BN2 -> fused scale/bias
// ---------------------------------------------------------------------------
__global__ void finalize2_kernel(const float* __restrict__ gamma2,
                                 const float* __restrict__ beta2,
                                 const float* __restrict__ gsum,
                                 const float* __restrict__ gsq,
                                 float* __restrict__ scale2, float* __restrict__ bias2) {
    int o = threadIdx.x;
    if (o >= OC) return;
    const float inv = 1.0f / (float)(NB * NPTS * KNN);
    float mean = gsum[o] * inv;
    float var = gsq[o] * inv - mean * mean;
    float sc = gamma2[o] * rsqrtf(var + EPSF);
    scale2[o] = sc;
    bias2[o] = beta2[o] - mean * sc;
}

// ---------------------------------------------------------------------------
// Kernel 6: final pass — recompute h1n, WMMA h2, BN2 + LeakyReLU, max over K,
// write out (B, 64, N).
// ---------------------------------------------------------------------------
__global__ void __launch_bounds__(256)
edgeconv_final_kernel(const float* __restrict__ x, const int* __restrict__ idx,
                      const float* __restrict__ W1f, const float* __restrict__ W2f,
                      const float* __restrict__ scale1, const float* __restrict__ bias1,
                      const float* __restrict__ scale2, const float* __restrict__ bias2,
                      float* __restrict__ out) {
    __shared__ _Float16 sW2[OC][APAD];
    __shared__ _Float16 sA[PPB][KNN][APAD];
    __shared__ float sW1[OC * 6];
    __shared__ float sS1[OC], sB1[OC];
    __shared__ float sS2[OC], sB2[OC];
    __shared__ float sOut[PPB][OC];

    const int tid = threadIdx.x;
    for (int i = tid; i < OC * OC; i += 256) sW2[i >> 6][i & 63] = (_Float16)W2f[i];
    for (int i = tid; i < OC * 6; i += 256) sW1[i] = W1f[i];
    if (tid < OC) {
        sS1[tid] = scale1[tid]; sB1[tid] = bias1[tid];
        sS2[tid] = scale2[tid]; sB2[tid] = bias2[tid];
    }
    __syncthreads();

    const int w = tid >> 5, lane = tid & 31;
    const int p = blockIdx.x * PPB + w;
    compute_h1_tile(p, lane, x, idx, sW1, sS1, sB1, &sA[w][lane][0]);
    __syncthreads();

    const int l16 = lane & 15;
    const bool hi = lane >= 16;
    v8f acc[2][4] = {};
    WMMA_CORE(sA[w], sW2, acc, l16, hi);

    #pragma unroll
    for (int nt = 0; nt < 4; ++nt) {
        const int o = nt * 16 + l16;
        const float sc = sS2[o], bi = sB2[o];
        float m = -3.4e38f;
        #pragma unroll
        for (int i = 0; i < 8; ++i) {
            float v0 = acc[0][nt][i] * sc + bi; v0 = (v0 >= 0.f) ? v0 : SLOPE * v0;
            float v1 = acc[1][nt][i] * sc + bi; v1 = (v1 >= 0.f) ? v1 : SLOPE * v1;
            m = fmaxf(m, fmaxf(v0, v1));
        }
        m = fmaxf(m, __shfl_xor(m, 16, 32));   // merge the two M half-lanes
        if (!hi) sOut[w][o] = m;
    }
    __syncthreads();

    // coalesced-ish write: consecutive tid -> consecutive n for a fixed channel
    for (int q = tid; q < PPB * OC; q += 256) {
        int o = q >> 3, pp = q & 7;
        int pt = blockIdx.x * PPB + pp;
        int b = pt >> 12, n = pt & (NPTS - 1);
        out[((size_t)(b * OC + o)) * NPTS + n] = sOut[pp][o];
    }
}

// ---------------------------------------------------------------------------
extern "C" void kernel_launch(void* const* d_in, const int* in_sizes, int n_in,
                              void* d_out, int out_size, void* d_ws, size_t ws_size,
                              hipStream_t stream) {
    const float* x  = (const float*)d_in[0];
    const float* W1 = (const float*)d_in[1];
    const float* g1 = (const float*)d_in[2];
    const float* b1 = (const float*)d_in[3];
    const float* W2 = (const float*)d_in[4];
    const float* g2 = (const float*)d_in[5];
    const float* b2 = (const float*)d_in[6];
    float* out = (float*)d_out;

    int*   idx = (int*)d_ws;                                   // B*N*K ints = 4 MB
    float* wsf = (float*)((char*)d_ws + (size_t)NB * NPTS * KNN * sizeof(int));
    float* mom    = wsf;         // 42 (padded to 48)
    float* gsum   = wsf + 48;    // 64
    float* gsq    = wsf + 112;   // 64
    float* scale1 = wsf + 176;
    float* bias1  = wsf + 240;
    float* scale2 = wsf + 304;
    float* bias2  = wsf + 368;

    hipMemsetAsync(wsf, 0, 176 * sizeof(float), stream);       // zero accumulators

    knn_kernel<<<NB * NPTS, 256, 0, stream>>>(x, idx);
    moments_kernel<<<256, 256, 0, stream>>>(x, idx, mom);
    finalize1_kernel<<<1, 64, 0, stream>>>(W1, g1, b1, mom, scale1, bias1);
    stats2_kernel<<<NB * NPTS / PPB, 256, 0, stream>>>(x, idx, W1, W2, scale1, bias1, gsum, gsq);
    finalize2_kernel<<<1, 64, 0, stream>>>(g2, b2, gsum, gsq, scale2, bias2);
    edgeconv_final_kernel<<<NB * NPTS / PPB, 256, 0, stream>>>(x, idx, W1, W2, scale1, bias1,
                                                               scale2, bias2, out);
}
